// LSTM_20452634264146
// MI455X (gfx1250) — matrix operand order repaired and checked by reading, not compiled, not run on hardware
//
#include <hip/hip_runtime.h>
#include <hip/hip_bf16.h>

#define SEQ   512
#define EMBD  1024
#define HID   2048
#define NCLS  10
#define BATCH 256
#define NSTEP (SEQ - 1)

typedef __attribute__((ext_vector_type(16))) __bf16 v16bf;
typedef __attribute__((ext_vector_type(8)))  float  v8f;

// ---------------- helpers ----------------

__device__ __forceinline__ unsigned short f32_to_bf16(float f) {
  union { float f; unsigned int u; } v; v.f = f;
  unsigned int u = v.u;
  if ((u & 0x7F800000u) == 0x7F800000u) return (unsigned short)(u >> 16); // inf/nan: truncate
  return (unsigned short)((u + 0x7FFFu + ((u >> 16) & 1u)) >> 16);        // RNE
}

__device__ __forceinline__ float bf16_to_f32(unsigned short h) {
  union { float f; unsigned int u; } v; v.u = ((unsigned int)h) << 16;
  return v.f;
}

// Fast activations: single v_exp_f32 + v_rcp_f32 each.
__device__ __forceinline__ float fast_tanh(float v) {
  const float e = __expf(2.f * v);
  return 1.f - 2.f * __builtin_amdgcn_rcpf(e + 1.f);   // (e-1)/(e+1)
}
__device__ __forceinline__ float fast_sigmoid(float v) {
  return __builtin_amdgcn_rcpf(1.f + __expf(-v));
}

// A-fragment (16-bit A 16x32): lanes 0-15 hold K 0..15, lanes 16-31 hold K 16..31 (contiguous 32B).
// p points at (row base + k + 16*khalf).
__device__ __forceinline__ v16bf load_frag_a(const unsigned short* p) {
  union { v16bf v; uint4 q[2]; } u;
  u.q[0] = *(const uint4*)(p);
  u.q[1] = *(const uint4*)(p + 8);
  return u.v;
}

// B-fragment (32x16): lanes 0-15 hold K 0..15, lanes 16-31 hold K 16..31 (contiguous 32B).
// p points at (col base + k + 16*khalf).
__device__ __forceinline__ v16bf load_frag_b(const unsigned short* p) {
  union { v16bf v; uint4 q[2]; } u;
  u.q[0] = *(const uint4*)(p);
  u.q[1] = *(const uint4*)(p + 8);
  return u.v;
}

// Device-wide split barrier: monotonic counter, one signaller per block.
__device__ __forceinline__ void grid_sync(unsigned int* cnt, unsigned int& target) {
  __threadfence();                       // release this block's writes
  __syncthreads();
  if (threadIdx.x == 0) {
    target += gridDim.x;
    __hip_atomic_fetch_add(cnt, 1u, __ATOMIC_RELEASE, __HIP_MEMORY_SCOPE_AGENT);
    while (__hip_atomic_load(cnt, __ATOMIC_ACQUIRE, __HIP_MEMORY_SCOPE_AGENT) < target) {
      __builtin_amdgcn_s_sleep(8);
    }
  }
  __syncthreads();
  __threadfence();                       // acquire other blocks' writes
}

// ---------------- prep kernels ----------------

__global__ void zero_u32_kernel(unsigned int* __restrict__ p, int n) {
  int i = blockIdx.x * blockDim.x + threadIdx.x;
  if (i < n) p[i] = 0u;
}

__global__ void to_bf16_kernel(const float* __restrict__ src,
                               unsigned short* __restrict__ dst, int n) {
  int i = blockIdx.x * blockDim.x + threadIdx.x;
  if (i < n) dst[i] = f32_to_bf16(src[i]);
}

// src: [K][N] f32 row-major  ->  dst: [N][K] bf16 row-major (LDS-tiled transpose)
__global__ void transpose_bf16_kernel(const float* __restrict__ src,
                                      unsigned short* __restrict__ dst,
                                      int K, int N) {
  __shared__ float tile[32][33];
  const int kb = blockIdx.y * 32;
  const int nb = blockIdx.x * 32;
  const int tx = threadIdx.x & 31;
  const int ty = threadIdx.x >> 5;
#pragma unroll
  for (int r = ty; r < 32; r += 8)
    tile[r][tx] = src[(size_t)(kb + r) * N + (nb + tx)];
  __syncthreads();
#pragma unroll
  for (int r = ty; r < 32; r += 8)
    dst[(size_t)(nb + r) * K + (kb + tx)] = f32_to_bf16(tile[tx][r]);
}

// ---------------- E = emb @ Wx + b  (both gates, one pass; direct-from-L2) ----------------

template <int KDIM>
__device__ __forceinline__ void dual_gemm_tile(
    const unsigned short* __restrict__ A,
    const unsigned short* __restrict__ B0t,
    const unsigned short* __restrict__ B1t,
    int mBase, int nBase, int lane,
    v8f acc0[2][4], v8f acc1[2][4])
{
  const int mlo   = lane & 15;
  const int khalf = lane >> 4;

  const unsigned short* aRow0 = A + (size_t)(mBase + mlo) * KDIM;
  const unsigned short* aRow1 = A + (size_t)(mBase + 16 + mlo) * KDIM;
  const unsigned short* b0Rows[4];
  const unsigned short* b1Rows[4];
#pragma unroll
  for (int nt = 0; nt < 4; ++nt) {
    b0Rows[nt] = B0t + (size_t)(nBase + nt * 16 + mlo) * KDIM;
    b1Rows[nt] = B1t + (size_t)(nBase + nt * 16 + mlo) * KDIM;
  }
  for (int k = 0; k < KDIM; k += 32) {
    const int offA = k + khalf * 16;
    const int offB = k + khalf * 16;
    v16bf a0 = load_frag_a(aRow0 + offA);
    v16bf a1 = load_frag_a(aRow1 + offA);
#pragma unroll
    for (int nt = 0; nt < 4; ++nt) {
      v16bf b0 = load_frag_b(b0Rows[nt] + offB);
      acc0[0][nt] = __builtin_amdgcn_wmma_f32_16x16x32_bf16(false, a0, false, b0,
                                                            (short)0, acc0[0][nt], false, false);
      acc0[1][nt] = __builtin_amdgcn_wmma_f32_16x16x32_bf16(false, a1, false, b0,
                                                            (short)0, acc0[1][nt], false, false);
      v16bf b1 = load_frag_b(b1Rows[nt] + offB);
      acc1[0][nt] = __builtin_amdgcn_wmma_f32_16x16x32_bf16(false, a0, false, b1,
                                                            (short)0, acc1[0][nt], false, false);
      acc1[1][nt] = __builtin_amdgcn_wmma_f32_16x16x32_bf16(false, a1, false, b1,
                                                            (short)0, acc1[1][nt], false, false);
    }
  }
}

__global__ void __launch_bounds__(32)
egemm_kernel(const unsigned short* __restrict__ embB,   // [SEQ][EMBD] bf16
             const unsigned short* __restrict__ WgxT,   // [HID][EMBD] bf16
             const unsigned short* __restrict__ WixT,   // [HID][EMBD] bf16
             const float* __restrict__ bg, const float* __restrict__ bi,
             float* __restrict__ Eg, float* __restrict__ Ei)  // [SEQ][HID]
{
  const int lane  = threadIdx.x;
  const int nBase = blockIdx.x * 64;
  const int mBase = blockIdx.y * 32;

  v8f accg[2][4], acci[2][4];
#pragma unroll
  for (int a = 0; a < 2; ++a)
#pragma unroll
    for (int b = 0; b < 4; ++b) { accg[a][b] = (v8f)0.f; acci[a][b] = (v8f)0.f; }

  dual_gemm_tile<EMBD>(embB, WgxT, WixT, mBase, nBase, lane, accg, acci);

  const int nlo = lane & 15;
  const int mhi = (lane >> 4) * 8;
#pragma unroll
  for (int mt = 0; mt < 2; ++mt)
#pragma unroll
    for (int nt = 0; nt < 4; ++nt)
#pragma unroll
      for (int r = 0; r < 8; ++r) {
        const int m = mBase + 16 * mt + r + mhi;
        const int n = nBase + 16 * nt + nlo;
        Eg[(size_t)m * HID + n] = accg[mt][nt][r] + bg[n];
        Ei[(size_t)m * HID + n] = acci[mt][nt][r] + bi[n];
      }
}

// ---------------- persistent recurrent kernel ----------------
//
// 128 blocks x 256 threads (8 waves).  Block owns a 16-column N-slice for the
// FULL K (no split-K): its dual-gate weight slice [2][16][2048] bf16 = 128 KB
// is pinned in LDS for all 511 steps (<=2 blocks/WGP at 320 KB LDS, so the
// grid co-resides even on a 64-WGP part).  Wave w covers batch rows
// [32w, 32w+32); each lane owns 16 fixed (b, n) cells whose c-state lives in
// REGISTERS for the whole recurrence.  One grid sync per step; h ping-pongs
// between two global buffers so writers never race readers.
__global__ void __launch_bounds__(256)
lstm_persistent(unsigned short* __restrict__ hA,        // [BATCH][HID] bf16
                unsigned short* __restrict__ hB,        // [BATCH][HID] bf16
                const int* __restrict__ x,              // [BATCH][SEQ]
                const unsigned short* __restrict__ WghT,// [HID][HID] bf16
                const unsigned short* __restrict__ WihT,// [HID][HID] bf16
                const float* __restrict__ Eg,           // [SEQ][HID]
                const float* __restrict__ Ei,
                unsigned int* __restrict__ barCnt)
{
  extern __shared__ unsigned short smemW[];   // [2 gates][16 cols][2048 k] bf16 = 128 KB

  const int tid   = threadIdx.x;
  const int lane  = tid & 31;
  const int wave  = tid >> 5;
  const int blk   = blockIdx.x;          // 0..127
  const int nBase = blk * 16;

  // ---- stage the weight slice into LDS once (256B per thread) ----
  {
    const int elem = tid * 256;          // 65536 elems total
    const int g    = elem >> 15;         // 16*2048 elems per gate
    const int rem  = elem & 32767;
    const int n    = rem >> 11;          // /2048
    const int k    = rem & 2047;
    const unsigned short* src =
        (g == 0 ? WghT : WihT) + (size_t)(nBase + n) * HID + k;
    unsigned short* dst = smemW + elem;
#pragma unroll
    for (int i = 0; i < 16; ++i)
      ((uint4*)dst)[i] = ((const uint4*)src)[i];
  }

  // ---- zero h0 (each block clears its 8KB share of hA) ----
  {
    unsigned int* hz = (unsigned int*)hA;              // BATCH*HID/2 u32
    const int per = (BATCH * HID / 2) / 128;           // 2048 u32 per block
    for (int i = tid; i < per; i += 256) hz[blk * per + i] = 0u;
  }

  unsigned int target = 0;
  grid_sync(barCnt, target);             // weights + h0 visible everywhere

  const int mlo   = lane & 15;
  const int khalf = lane >> 4;
  const int mBase = wave * 32;           // 8 waves cover all 256 batch rows
  const int nCol  = nBase + mlo;         // this lane's output column

  const unsigned short* bRowG = smemW + (size_t)mlo * HID;            // gate g
  const unsigned short* bRowI = smemW + (size_t)(16 + mlo) * HID;     // gate i

  float creg[16];                         // cell state lives in registers
#pragma unroll
  for (int i = 0; i < 16; ++i) creg[i] = 0.f;

  for (int t = 0; t < NSTEP; ++t) {
    const unsigned short* hIn  = (t & 1) ? hB : hA;
    unsigned short*       hOut = (t & 1) ? hA : hB;

    const unsigned short* aRow0 = hIn + (size_t)(mBase + mlo) * HID;
    const unsigned short* aRow1 = hIn + (size_t)(mBase + 16 + mlo) * HID;

    // ---------- dual-gate GEMM over the full K ----------
    v8f acc[2][2];                        // [gate][mt]
#pragma unroll
    for (int g = 0; g < 2; ++g)
#pragma unroll
      for (int mt = 0; mt < 2; ++mt) acc[g][mt] = (v8f)0.f;

    for (int c = 0; c < HID; c += 32) {
      const int offA = c + khalf * 16;
      const int offB = c + khalf * 16;
      v16bf a0 = load_frag_a(aRow0 + offA);
      v16bf a1 = load_frag_a(aRow1 + offA);
      v16bf bg_ = load_frag_b(bRowG + offB);
      acc[0][0] = __builtin_amdgcn_wmma_f32_16x16x32_bf16(false, a0, false, bg_,
                                                          (short)0, acc[0][0], false, false);
      acc[0][1] = __builtin_amdgcn_wmma_f32_16x16x32_bf16(false, a1, false, bg_,
                                                          (short)0, acc[0][1], false, false);
      v16bf bi_ = load_frag_b(bRowI + offB);
      acc[1][0] = __builtin_amdgcn_wmma_f32_16x16x32_bf16(false, a0, false, bi_,
                                                          (short)0, acc[1][0], false, false);
      acc[1][1] = __builtin_amdgcn_wmma_f32_16x16x32_bf16(false, a1, false, bi_,
                                                          (short)0, acc[1][1], false, false);
    }

    // ---------- pointwise epilogue on this lane's own cells ----------
#pragma unroll
    for (int mt = 0; mt < 2; ++mt)
#pragma unroll
      for (int r = 0; r < 8; ++r) {
        const int idx = mt * 8 + r;
        const int b   = mBase + 16 * mt + r + 8 * khalf;
        const int tok = x[b * SEQ + t];
        const float gp = acc[0][mt][r] + Eg[(size_t)tok * HID + nCol];
        const float ip = acc[1][mt][r] + Ei[(size_t)tok * HID + nCol];
        const float gv = fast_tanh(gp);
        const float iv = fast_sigmoid(ip);
        const float cv = fmaf(gv, iv, iv * creg[idx]);   // c = g*i + i*c
        creg[idx] = cv;
        hOut[(size_t)b * HID + nCol] = f32_to_bf16(fast_tanh(cv) * iv);
      }

    grid_sync(barCnt, target);           // h(t+1) published before next read
  }
}

// ---------------- final projection: out = h @ wp + bp ----------------

__global__ void proj_kernel(const unsigned short* __restrict__ h,  // [BATCH][HID] bf16
                            const float* __restrict__ wp,          // [HID][NCLS]
                            const float* __restrict__ bp,
                            float* __restrict__ out)               // [BATCH][NCLS]
{
  __shared__ float red[256 * NCLS];
  const int b   = blockIdx.x;
  const int tid = threadIdx.x;

  float acc[NCLS];
#pragma unroll
  for (int j = 0; j < NCLS; ++j) acc[j] = 0.f;

  for (int k = tid; k < HID; k += 256) {
    const float hv = bf16_to_f32(h[(size_t)b * HID + k]);
    const float* w = wp + (size_t)k * NCLS;
#pragma unroll
    for (int j = 0; j < NCLS; ++j) acc[j] += hv * w[j];
  }
#pragma unroll
  for (int j = 0; j < NCLS; ++j) red[tid * NCLS + j] = acc[j];
  __syncthreads();

  for (int s = 128; s > 0; s >>= 1) {
    if (tid < s)
#pragma unroll
      for (int j = 0; j < NCLS; ++j) red[tid * NCLS + j] += red[(tid + s) * NCLS + j];
    __syncthreads();
  }
  if (tid == 0)
#pragma unroll
    for (int j = 0; j < NCLS; ++j) out[b * NCLS + j] = red[j] + bp[j];
}

// ---------------- launch ----------------

extern "C" void kernel_launch(void* const* d_in, const int* in_sizes, int n_in,
                              void* d_out, int out_size, void* d_ws, size_t ws_size,
                              hipStream_t stream) {
  (void)in_sizes; (void)n_in; (void)out_size; (void)ws_size;

  const int*   x   = (const int*)d_in[0];
  const float* emb = (const float*)d_in[1];
  const float* wgx = (const float*)d_in[2];
  const float* wgh = (const float*)d_in[3];
  const float* bg  = (const float*)d_in[4];
  const float* wix = (const float*)d_in[5];
  const float* wih = (const float*)d_in[6];
  const float* bi  = (const float*)d_in[7];
  // d_in[8..13] (wfx,wfh,bf,wox,woh,bo) are dead code in the reference.
  const float* wp  = (const float*)d_in[14];
  const float* bp  = (const float*)d_in[15];
  float* out = (float*)d_out;

  // workspace layout (≈36 MB)
  char* base = (char*)d_ws;
  size_t off = 0;
  auto take = [&](size_t bytes) -> char* {
    char* r = base + off;
    off += bytes;
    off = (off + 255) & ~(size_t)255;
    return r;
  };
  unsigned short* embB = (unsigned short*)take((size_t)SEQ * EMBD * 2);
  unsigned short* WgxT = (unsigned short*)take((size_t)HID * EMBD * 2);
  unsigned short* WixT = (unsigned short*)take((size_t)HID * EMBD * 2);
  unsigned short* WghT = (unsigned short*)take((size_t)HID * HID * 2);
  unsigned short* WihT = (unsigned short*)take((size_t)HID * HID * 2);
  float* Eg   = (float*)take((size_t)SEQ * HID * 4);
  float* Ei   = (float*)take((size_t)SEQ * HID * 4);
  unsigned short* hA = (unsigned short*)take((size_t)BATCH * HID * 2);
  unsigned short* hB = (unsigned short*)take((size_t)BATCH * HID * 2);
  unsigned int* barCnt = (unsigned int*)take(256);

  // 1) conversions / transposes
  {
    int n = SEQ * EMBD;
    to_bf16_kernel<<<(n + 255) / 256, 256, 0, stream>>>(emb, embB, n);
  }
  transpose_bf16_kernel<<<dim3(HID / 32, EMBD / 32), 256, 0, stream>>>(wgx, WgxT, EMBD, HID);
  transpose_bf16_kernel<<<dim3(HID / 32, EMBD / 32), 256, 0, stream>>>(wix, WixT, EMBD, HID);
  transpose_bf16_kernel<<<dim3(HID / 32, HID / 32),  256, 0, stream>>>(wgh, WghT, HID, HID);
  transpose_bf16_kernel<<<dim3(HID / 32, HID / 32),  256, 0, stream>>>(wih, WihT, HID, HID);

  // 2) reset the device-wide barrier counter (graph replays re-run this)
  zero_u32_kernel<<<1, 32, 0, stream>>>(barCnt, 1);

  // 3) fold all input GEMMs into per-vocab-row tables: E = emb @ Wx + b
  egemm_kernel<<<dim3(HID / 64, SEQ / 32), 32, 0, stream>>>(embB, WgxT, WixT, bg, bi, Eg, Ei);

  // 4) one persistent kernel runs all 511 steps (weights pinned in LDS, c in regs,
  //    one grid sync per step, h ping-pong)
  lstm_persistent<<<128, 256, 128 * 1024, stream>>>(hA, hB, x, WghT, WihT, Eg, Ei, barCnt);

  // 5) projection: NSTEP = 511 (odd) -> final h is in hB
  proj_kernel<<<BATCH, 256, 0, stream>>>(hB, wp, bp, out);
}
